// SAModule_2860448219403
// MI455X (gfx1250) — matrix operand order repaired and checked by reading, not compiled, not run on hardware
//
#include <hip/hip_runtime.h>

// ---------------- problem constants ----------------
constexpr int NB   = 16;     // batch
constexpr int NPTS = 4096;   // N points
constexpr int NC   = 64;     // input channels
constexpr int NM   = 1024;   // sampled centers per cloud
constexpr int NK   = 64;     // max neighbors
constexpr int NH   = 128;    // MLP width
constexpr float R2   = 0.04f;   // radius^2 (R=0.2)
constexpr float NEGV = -1e30f;

constexpr int KP1 = 48;      // packed k-pair rows of W1 (96/2)
constexpr int KP2 = 64;      // packed k-pair rows of W2 (128/2)
constexpr int AS  = 104;     // LDS stride (halves) for A tile (96 + 8 pad)
constexpr int HS  = 136;     // LDS stride (halves) for H tile (128 + 8 pad)

typedef __attribute__((ext_vector_type(16))) _Float16 v16h;
typedef __attribute__((ext_vector_type(8)))  float    v8f;

union FragH {
    v16h v;
    unsigned int u[8];
    _Float16 h[16];
};

// ---------------- 1) FPS: one block per cloud ----------------
__global__ __launch_bounds__(1024)
void fps_kernel(const float* __restrict__ pos, float* __restrict__ pos_s)
{
    const int b = blockIdx.x;
    const int t = threadIdx.x;
    const float* pb = pos + (size_t)b * NPTS * 3;

    float px[4], py[4], pz[4], dd[4];
#pragma unroll
    for (int i = 0; i < 4; ++i) {
        int id = t * 4 + i;
        px[i] = pb[id * 3 + 0];
        py[i] = pb[id * 3 + 1];
        pz[i] = pb[id * 3 + 2];
        dd[i] = 3.4e38f;
    }

    __shared__ float lastS[3];
    __shared__ float redV[32];
    __shared__ int   redI[32];
    __shared__ int   sIdx;

    if (t == 0) {
        lastS[0] = pb[0]; lastS[1] = pb[1]; lastS[2] = pb[2];
        float* o = pos_s + (size_t)b * NM * 3;
        o[0] = pb[0]; o[1] = pb[1]; o[2] = pb[2];   // center 0 = point 0
    }
    __syncthreads();

    for (int it = 1; it < NM; ++it) {
        const float lx = lastS[0], ly = lastS[1], lz = lastS[2];
        float bv = -1.0f; int bi = 0;
#pragma unroll
        for (int i = 0; i < 4; ++i) {
            float dx = px[i] - lx, dy = py[i] - ly, dz = pz[i] - lz;
            float d = dx * dx + dy * dy + dz * dz;
            dd[i] = fminf(dd[i], d);
            if (dd[i] > bv) { bv = dd[i]; bi = t * 4 + i; }
        }
        for (int off = 16; off > 0; off >>= 1) {
            float ov = __shfl_xor(bv, off, 32);
            int   oi = __shfl_xor(bi, off, 32);
            if (ov > bv || (ov == bv && oi < bi)) { bv = ov; bi = oi; }
        }
        if ((t & 31) == 0) { redV[t >> 5] = bv; redI[t >> 5] = bi; }
        __syncthreads();
        if (t < 32) {
            bv = redV[t]; bi = redI[t];
            for (int off = 16; off > 0; off >>= 1) {
                float ov = __shfl_xor(bv, off, 32);
                int   oi = __shfl_xor(bi, off, 32);
                if (ov > bv || (ov == bv && oi < bi)) { bv = ov; bi = oi; }
            }
            if (t == 0) sIdx = bi;
        }
        __syncthreads();
        const int w = sIdx;
        if (t == (w >> 2)) {
            const int i = w & 3;
            lastS[0] = px[i]; lastS[1] = py[i]; lastS[2] = pz[i];
            float* o = pos_s + ((size_t)b * NM + it) * 3;
            o[0] = px[i]; o[1] = py[i]; o[2] = pz[i];
        }
        __syncthreads();
    }
}

// ---------------- 2) pack weights -> f16 k-pair-interleaved ----------------
__global__ __launch_bounds__(256)
void pack_kernel(const float* __restrict__ W1, const float* __restrict__ W2,
                 unsigned int* __restrict__ W1p, unsigned int* __restrict__ W2p)
{
    const int idx = blockIdx.x * 256 + threadIdx.x;
    const int row = idx >> 7;
    const int n   = idx & 127;
    union { _Float16 h[2]; unsigned int u; } pk;
    if (row < KP1) {
        const int k0 = row * 2, k1 = k0 + 1;
        float a = (k0 < NC + 3) ? W1[k0 * NH + n] : 0.0f;
        float b = (k1 < NC + 3) ? W1[k1 * NH + n] : 0.0f;
        pk.h[0] = (_Float16)a; pk.h[1] = (_Float16)b;
        W1p[row * NH + n] = pk.u;
    } else {
        const int r = row - KP1;
        const int k0 = r * 2;
        pk.h[0] = (_Float16)W2[k0 * NH + n];
        pk.h[1] = (_Float16)W2[(k0 + 1) * NH + n];
        W2p[r * NH + n] = pk.u;
    }
}

// ---------------- 3) ball query: one wave32 per center ----------------
__global__ __launch_bounds__(256)
void ball_kernel(const float* __restrict__ pos, const float* __restrict__ pos_s,
                 unsigned short* __restrict__ nbr, int* __restrict__ counts)
{
    const int lane = threadIdx.x & 31;
    const int cid  = blockIdx.x * 8 + (threadIdx.x >> 5);
    const int b    = cid >> 10;
    const float cx = pos_s[cid * 3 + 0];
    const float cy = pos_s[cid * 3 + 1];
    const float cz = pos_s[cid * 3 + 2];
    const float* pb = pos + (size_t)b * NPTS * 3;

    for (int s = lane; s < NK; s += 32) nbr[(size_t)cid * NK + s] = 0;

    int cnt = 0;
    for (int n0 = 0; n0 < NPTS && cnt < NK; n0 += 32) {
        const int n = n0 + lane;
        float dx = pb[n * 3 + 0] - cx;
        float dy = pb[n * 3 + 1] - cy;
        float dz = pb[n * 3 + 2] - cz;
        bool in = (dx * dx + dy * dy + dz * dz) <= R2;
        unsigned mask = (unsigned)__ballot(in);
        int rank = cnt + __popc(mask & ((1u << lane) - 1u));
        if (in && rank < NK) nbr[(size_t)cid * NK + rank] = (unsigned short)n;
        cnt += __popc(mask);
    }
    if (lane == 0) counts[cid] = cnt < NK ? cnt : NK;
}

// ---------------- 4) WMMA MLP + masked max: 8 centers per workgroup ----------------
// Async global->LDS double-buffered gather of x rows, f32->f16 convert pass,
// then two WMMA GEMMs with register-resident weight fragments.

__device__ __forceinline__ void issue_gather_async(
    const float* __restrict__ x, const unsigned short* __restrict__ nbr,
    int cid, int b, float* dstbase /* __shared__ */, int t)
{
    const int j0  = t >> 4;     // base row for this thread (rows j0 + 16*i)
    const int seg = t & 15;     // 16-byte segment within the 64-float row

    // batch the 4 neighbor-index loads (independent -> single load-wait)
    int njs[4];
#pragma unroll
    for (int i = 0; i < 4; ++i)
        njs[i] = (int)nbr[(size_t)cid * NK + j0 + 16 * i];

#pragma unroll
    for (int i = 0; i < 4; ++i) {
        const int j = j0 + 16 * i;
        const float* src = x + ((size_t)b * NPTS + njs[i]) * NC + seg * 4;
        unsigned ldsa = (unsigned)(size_t)(dstbase + j * NC + seg * 4);
        unsigned long long ga = (unsigned long long)(size_t)src;
        // CDNA5 async copy: 16B global -> LDS, tracked by ASYNCcnt
        asm volatile("global_load_async_to_lds_b128 %0, %1, off"
                     :: "v"(ldsa), "v"(ga) : "memory");
    }
}

__global__ __launch_bounds__(256)
void mlp_kernel(const float* __restrict__ x, const float* __restrict__ pos,
                const float* __restrict__ pos_s,
                const unsigned short* __restrict__ nbr, const int* __restrict__ counts,
                const float* __restrict__ b1, const float* __restrict__ b2,
                const unsigned int* __restrict__ W1p, const unsigned int* __restrict__ W2p,
                float* __restrict__ out)
{
    __shared__ float    Xs[2][64 * NC];   // async staging, double buffered (32 KB)
    __shared__ _Float16 A1[64 * AS];      // f16 feature tile
    __shared__ _Float16 Hs[64 * HS];      // f16 hidden tile

    const int t    = threadIdx.x;
    const int lane = t & 31;
    const int wv   = t >> 5;              // wave id = output column block
    const int half = lane >> 4;
    const int l16  = lane & 15;
    const int ncol = wv * 16 + l16;

    __builtin_prefetch(W1p + t, 0, 0);    // global_prefetch_b8
    __builtin_prefetch(W2p + t, 0, 0);

    const float b1v = b1[ncol];
    const float b2v = b2[ncol];

    // Register-resident B fragments (this code's contiguous k order, same as A):
    // lanes 0-15 -> K 0..15, lanes 16-31 -> K 16..31; VGPR i packs (K, K+1).
    FragH w1f[3];
#pragma unroll
    for (int kb = 0; kb < 3; ++kb)
#pragma unroll
        for (int i = 0; i < 8; ++i)
            w1f[kb].u[i] = W1p[(kb * 16 + half * 8 + i) * NH + ncol];
    FragH w2f[4];
#pragma unroll
    for (int kb = 0; kb < 4; ++kb)
#pragma unroll
        for (int i = 0; i < 8; ++i)
            w2f[kb].u[i] = W2p[(kb * 16 + half * 8 + i) * NH + ncol];

    // prologue: start async gather for first center
    {
        const int cid0 = blockIdx.x * 8;
        issue_gather_async(x, nbr, cid0, cid0 >> 10, &Xs[0][0], t);
    }

    for (int c = 0; c < 8; ++c) {
        const int cid = blockIdx.x * 8 + c;
        const int b   = cid >> 10;
        const int cnt = counts[cid];

        // all of this wave's async copies landed in LDS
        asm volatile("s_wait_asynccnt 0x0" ::: "memory");
        __syncthreads();   // + everyone else's; A1 also free from prev center

        {   // convert pass: Xs[c&1] (f32) -> A1 (f16), plus rel-pos / zero pad
            const int j = t >> 2;
            const int q = t & 3;
            const float* srow = &Xs[c & 1][j * NC + q * 16];
            float tmp[16];
            *(float4*)(tmp +  0) = ((const float4*)srow)[0];
            *(float4*)(tmp +  4) = ((const float4*)srow)[1];
            *(float4*)(tmp +  8) = ((const float4*)srow)[2];
            *(float4*)(tmp + 12) = ((const float4*)srow)[3];
            union { _Float16 h[16]; uint4 q4[2]; } pk;
#pragma unroll
            for (int i = 0; i < 16; ++i) pk.h[i] = (_Float16)tmp[i];
            uint4* dst4 = (uint4*)(A1 + (size_t)j * AS + q * 16);
            dst4[0] = pk.q4[0];
            dst4[1] = pk.q4[1];

            if (q == 3) {   // columns 64..95: rel-pos then zero pad
                const int nj = (int)nbr[(size_t)cid * NK + j];
                const float* pr = pos + ((size_t)b * NPTS + nj) * 3;
                const float* ps = pos_s + (size_t)cid * 3;
                _Float16* d = A1 + (size_t)j * AS + 64;
                uint4 z; z.x = 0u; z.y = 0u; z.z = 0u; z.w = 0u;
                ((uint4*)d)[0] = z; ((uint4*)d)[1] = z;
                ((uint4*)d)[2] = z; ((uint4*)d)[3] = z;
                d[0] = (_Float16)(pr[0] - ps[0]);
                d[1] = (_Float16)(pr[1] - ps[1]);
                d[2] = (_Float16)(pr[2] - ps[2]);
            }
        }
        __syncthreads();

        // prefetch next center's rows while we do the GEMMs
        if (c + 1 < 8)
            issue_gather_async(x, nbr, cid + 1, (cid + 1) >> 10,
                               &Xs[(c + 1) & 1][0], t);

        // ---- GEMM1: [64 x 96] f16 x [96 x 16] f16 -> relu -> Hs ----
#pragma unroll
        for (int rb = 0; rb < 4; ++rb) {
            v8f acc = {};
#pragma unroll
            for (int kb = 0; kb < 3; ++kb) {
                FragH af;
                const _Float16* arow = A1 + (rb * 16 + l16) * AS;
#pragma unroll
                for (int i = 0; i < 8; ++i) {
                    const int koff = kb * 32 + half * 16 + 2 * i;
                    af.u[i] = *(const unsigned int*)(arow + koff);
                }
                acc = __builtin_amdgcn_wmma_f32_16x16x32_f16(
                    false, af.v, false, w1f[kb].v, (short)0, acc, false, false);
            }
#pragma unroll
            for (int vv = 0; vv < 8; ++vv) {
                float hv = acc[vv] + b1v;
                hv = hv > 0.0f ? hv : 0.0f;
                const int r = rb * 16 + half * 8 + vv;
                Hs[r * HS + ncol] = (_Float16)hv;
            }
        }
        __syncthreads();

        // ---- GEMM2: [64 x 128] x [128 x 16] -> relu -> masked row max ----
        float cmax = NEGV;
#pragma unroll
        for (int rb = 0; rb < 4; ++rb) {
            v8f acc = {};
#pragma unroll
            for (int kb = 0; kb < 4; ++kb) {
                FragH af;
                const _Float16* arow = Hs + (rb * 16 + l16) * HS;
#pragma unroll
                for (int i = 0; i < 8; ++i) {
                    const int koff = kb * 32 + half * 16 + 2 * i;
                    af.u[i] = *(const unsigned int*)(arow + koff);
                }
                acc = __builtin_amdgcn_wmma_f32_16x16x32_f16(
                    false, af.v, false, w2f[kb].v, (short)0, acc, false, false);
            }
#pragma unroll
            for (int vv = 0; vv < 8; ++vv) {
                const int r = rb * 16 + half * 8 + vv;
                float hv = acc[vv] + b2v;
                hv = hv > 0.0f ? hv : 0.0f;
                if (r >= cnt) hv = NEGV;
                cmax = fmaxf(cmax, hv);
            }
        }
        const float om = fmaxf(cmax, __shfl_xor(cmax, 16, 32));
        if (half == 0)
            out[(size_t)cid * NH + ncol] = (cnt > 0) ? om : 0.0f;
    }
}

// ---------------- launcher ----------------
extern "C" void kernel_launch(void* const* d_in, const int* in_sizes, int n_in,
                              void* d_out, int out_size, void* d_ws, size_t ws_size,
                              hipStream_t stream)
{
    const float* x   = (const float*)d_in[0];
    const float* pos = (const float*)d_in[1];
    const float* W1  = (const float*)d_in[2];
    const float* b1  = (const float*)d_in[3];
    const float* W2  = (const float*)d_in[4];
    const float* b2  = (const float*)d_in[5];

    float* out   = (float*)d_out;                       // [B*M, 128]
    float* pos_s = out + (size_t)NB * NM * NH;          // [B*M, 3] (tuple tail)

    char* ws = (char*)d_ws;
    unsigned short* nbr    = (unsigned short*)ws;                    // 2 MB
    int*            counts = (int*)(ws + (size_t)NB * NM * NK * 2);  // 64 KB
    unsigned int*   W1p    = (unsigned int*)((char*)counts + (size_t)NB * NM * 4);
    unsigned int*   W2p    = W1p + KP1 * NH;

    fps_kernel <<<NB, 1024, 0, stream>>>(pos, pos_s);
    pack_kernel<<<((KP1 + KP2) * NH) / 256, 256, 0, stream>>>(W1, W2, W1p, W2p);
    ball_kernel<<<(NB * NM) / 8, 256, 0, stream>>>(pos, pos_s, nbr, counts);
    mlp_kernel <<<(NB * NM) / 8, 256, 0, stream>>>(x, pos, pos_s, nbr, counts,
                                                   b1, b2, W1p, W2p, out);
}
